// LinearAttention_31894427140801
// MI455X (gfx1250) — matrix operand order repaired and checked by reading, not compiled, not run on hardware
//
#include <hip/hip_runtime.h>
#include <hip/hip_bf16.h>

// ---------------------------------------------------------------------------
// LinearAttention for MI455X (gfx1250, wave32, WMMA).
// All matmuls via v_wmma_f32_16x16x32_bf16 (fp32 accum).
// B operands staged per-block into LDS with the CDNA5 async-to-LDS path
// (ASYNCcnt / s_wait_asynccnt), double-buffered against the WMMA stream.
// B fragments batch-loaded (4 independent ds_load_b128 pairs per group) so
// the scheduler overlaps LDS reads with the WMMA chain.
// Memory branch algebraically folded: mem_out = c * num.
// ---------------------------------------------------------------------------

#define DEVINL __device__ __forceinline__

typedef __bf16 bf16;
typedef __attribute__((ext_vector_type(16))) __bf16 v16bf;
typedef __attribute__((ext_vector_type(8)))  __bf16 v8bf;
typedef __attribute__((ext_vector_type(8)))  float  v8f;
typedef __attribute__((ext_vector_type(4)))  int    v4i;

static constexpr int   Bb   = 4;
static constexpr int   Ss   = 2048;
static constexpr int   Hh   = 1024;
static constexpr int   Nh   = 16;
static constexpr int   Dd   = 64;
static constexpr int   TOK  = Bb * Ss;         // 8192
static constexpr int   HH2  = Hh * Hh;         // one weight slice (elements)
static constexpr float EPSc = 1e-6f;

// ------------------- async global->LDS copy (16B / lane) -------------------
typedef __attribute__((address_space(1))) v4i g_v4i;
typedef __attribute__((address_space(3))) v4i l_v4i;

DEVINL void async_cp16(const bf16* __restrict__ g, bf16* l) {
#if __has_builtin(__builtin_amdgcn_global_load_async_to_lds_b128)
  __builtin_amdgcn_global_load_async_to_lds_b128((g_v4i*)g, (l_v4i*)l, 0, 0);
#else
  *(v8bf*)l = *(const v8bf*)g;                 // fallback: vmem load + ds_store
#endif
}

DEVINL void wait_async() {
#if __has_builtin(__builtin_amdgcn_s_wait_asynccnt)
  __builtin_amdgcn_s_wait_asynccnt(0);
#else
  asm volatile("s_wait_asynccnt 0" ::: "memory");
#endif
}

// ---------------- WMMA fragment helpers (ISA 7.12.2 layouts) ---------------
DEVINL v16bf frag_a(const bf16* __restrict__ p, int lda) {
  const int lane = threadIdx.x & 31;
  const bf16* q  = p + (size_t)(lane & 15) * lda + (lane >> 4) * 8;
  union { v16bf v; v8bf h[2]; } u;
  u.h[0] = *(const v8bf*)(q);
  u.h[1] = *(const v8bf*)(q + 16);
  return u.v;
}

DEVINL v16bf frag_b_lds(const bf16* pT, int ldk) {
  const int lane = threadIdx.x & 31;
  const bf16* q  = pT + (size_t)(lane & 15) * ldk + (lane >> 4) * 8;
  union { v16bf v; v8bf h[2]; } u;
  u.h[0] = *(const v8bf*)(q);
  u.h[1] = *(const v8bf*)(q + 16);
  return u.v;
}

DEVINL v8f wmma_bf16(v16bf a, v16bf b, v8f c) {
  return __builtin_amdgcn_wmma_f32_16x16x32_bf16(
      false, a, false, b, (short)0, c, false, false);
}

// Stage one 64(N) x 32(K) bf16 B-tile into LDS. 128 threads x 2 x 16B.
DEVINL void stage_btile(const bf16* __restrict__ gsrc, int ld, bf16* lds) {
#pragma unroll
  for (int c = 0; c < 2; ++c) {
    const int ci   = threadIdx.x + 128 * c;    // 0..255
    const int row  = ci >> 2;
    const int part = ci & 3;
    async_cp16(gsrc + (size_t)row * ld + part * 8, lds + row * 32 + part * 8);
  }
}

// Load 4 n-tile B fragments into independent registers, then 4 WMMAs.
DEVINL void mma_group(v16bf a, const bf16* ltile, v8f acc[4]) {
  v16bf f[4];
#pragma unroll
  for (int nt = 0; nt < 4; ++nt) f[nt] = frag_b_lds(ltile + nt * 16 * 32, 32);
#pragma unroll
  for (int nt = 0; nt < 4; ++nt) acc[nt] = wmma_bf16(a, f[nt], acc[nt]);
}

DEVINL float sigmoidf_(float x) { return 1.f / (1.f + __expf(-x)); }
DEVINL float kernel_elu(float x) {
  return (x > 0.f ? x : __expf(x) - 1.f) + 1.0f + EPSc;
}

// ------------------------- conversion kernels ------------------------------
__global__ void la_cvt_f32_bf16(const float* __restrict__ in,
                                bf16* __restrict__ out, int n) {
  int t = blockIdx.x * blockDim.x + threadIdx.x;
  if (t < n) out[t] = (bf16)in[t];
}

// W (K x N row-major fp32) -> WT (N x K row-major bf16)
__global__ void la_cvt_transpose_w(const float* __restrict__ W,
                                   bf16* __restrict__ WT) {
  int t = blockIdx.x * blockDim.x + threadIdx.x;   // t = k*1024 + n
  int k = t >> 10, n = t & 1023;
  WT[(size_t)n * Hh + k] = (bf16)W[t];
}

// --------------------- fused QKV projection (WMMA) -------------------------
// Wcat = [WqT | WkT | WvT] contiguous: single base pointer, 2MB slice
// offsets fold into 24-bit instruction offsets (no pointer spills).
__global__ void la_qkv_proj(const bf16* __restrict__ xb,
                            const bf16* __restrict__ Wcat,
                            const float* __restrict__ bq,
                            const float* __restrict__ bk,
                            const float* __restrict__ bv,
                            bf16* __restrict__ qo,
                            bf16* __restrict__ ko,
                            bf16* __restrict__ vo) {
  __shared__ bf16 lb[2][3][64 * 32];             // 24 KB
  const int n0   = blockIdx.x * 64;
  const int wave = threadIdx.x >> 5;
  const int lane = threadIdx.x & 31;
  const int m0   = blockIdx.y * 64 + wave * 16;

  const bf16* wT   = Wcat + (size_t)n0 * Hh;     // + w*HH2 folds as immediate
  const bf16* arow = xb + (size_t)m0 * Hh;
  v8f accq[4] = {}, acck[4] = {}, accv[4] = {};

#pragma unroll
  for (int w = 0; w < 3; ++w) stage_btile(wT + (size_t)w * HH2, Hh, lb[0][w]);

  for (int k0 = 0; k0 < Hh; k0 += 32) {
    const int cur = (k0 >> 5) & 1;
    wait_async();
    __syncthreads();                              // buf[cur] ready for all waves
    if (k0 + 32 < Hh) {
#pragma unroll
      for (int w = 0; w < 3; ++w)
        stage_btile(wT + (size_t)w * HH2 + k0 + 32, Hh, lb[cur ^ 1][w]);
    }
    __builtin_prefetch(arow + k0 + 64, 0, 1);     // global_prefetch_b8 (A)
    v16bf a = frag_a(arow + k0, Hh);
    mma_group(a, lb[cur][0], accq);
    mma_group(a, lb[cur][1], acck);
    mma_group(a, lb[cur][2], accv);
    __syncthreads();                              // reads done before rewrite
  }
  const int rbase = m0 + (lane >> 4) * 8;
  const int cidx  = lane & 15;
#pragma unroll
  for (int nt = 0; nt < 4; ++nt) {
    const int col = n0 + nt * 16 + cidx;
#pragma unroll
    for (int i = 0; i < 8; ++i) {
      const size_t off = (size_t)(rbase + i) * Hh + col;
      qo[off] = (bf16)(accq[nt][i] + bq[col]);
      ko[off] = (bf16)(acck[nt][i] + bk[col]);
      vo[off] = (bf16)(accv[nt][i] + bv[col]);
    }
  }
}

// ------------- RoPE + elu-kernel + mask; emits K^T and V^T -----------------
__global__ void la_rope_mask(bf16* __restrict__ q,
                             const bf16* __restrict__ kin,
                             const bf16* __restrict__ vin,
                             bf16* __restrict__ kt,
                             bf16* __restrict__ vt,
                             const float* __restrict__ mask) {
  const int t   = blockIdx.x * blockDim.x + threadIdx.x;
  const int j   = t & 31;
  const int n   = (t >> 5) & 15;
  const int tok = t >> 9;
  const int b   = tok >> 11;
  const int s   = tok & 2047;

  const size_t base = (size_t)tok * Hh + n * Dd;
  float q1 = (float)q[base + j],   q2 = (float)q[base + j + 32];
  float k1 = (float)kin[base + j], k2 = (float)kin[base + j + 32];

  const float invf = __expf(-(2.0f * (float)j / 64.0f) * __logf(10000.0f));
  float sn, cs;
  __sincosf((float)s * invf, &sn, &cs);

  float qr1 = q1 * cs - q2 * sn, qr2 = q2 * cs + q1 * sn;
  float kr1 = k1 * cs - k2 * sn, kr2 = k2 * cs + k1 * sn;

  const float m = mask[tok];
  const float scaling = 0.125f;                 // D^-0.5
  qr1 = kernel_elu(qr1) * scaling;
  qr2 = kernel_elu(qr2) * scaling;
  kr1 = kernel_elu(kr1) * m;
  kr2 = kernel_elu(kr2) * m;

  q[base + j]      = (bf16)qr1;
  q[base + j + 32] = (bf16)qr2;

  const size_t kvb0 = ((size_t)b * Nh + n) * Dd * Ss;
  kt[kvb0 + (size_t)j * Ss + s]        = (bf16)kr1;
  kt[kvb0 + (size_t)(j + 32) * Ss + s] = (bf16)kr2;
  vt[kvb0 + (size_t)j * Ss + s]        = (bf16)((float)vin[base + j] * m);
  vt[kvb0 + (size_t)(j + 32) * Ss + s] = (bf16)((float)vin[base + j + 32] * m);
}

// ----------------------- gate GEMV: g = sigmoid(x @ Wg + bg) ---------------
__global__ void la_gate(const float* __restrict__ x,
                        const float* __restrict__ Wg,
                        const float* __restrict__ bg,
                        float* __restrict__ gate) {
  const int wave = threadIdx.x >> 5;
  const int lane = threadIdx.x & 31;
  const int tok  = blockIdx.x * (blockDim.x >> 5) + wave;
  const float* row = x + (size_t)tok * Hh;
  float sum = 0.f;
  for (int i = lane; i < Hh; i += 32) sum += row[i] * Wg[i];
#pragma unroll
  for (int off = 16; off > 0; off >>= 1) sum += __shfl_xor(sum, off, 32);
  if (lane == 0) gate[tok] = sigmoidf_(sum + bg[0]);
}

// ----------- kv[b,n] = K^T V : 64x64x2048 GEMM per (b,n), WMMA ------------
__global__ void la_kv(const bf16* __restrict__ kt,
                      const bf16* __restrict__ vt,
                      float* __restrict__ kv) {
  __shared__ bf16 lb[2][64 * 32];                // 8 KB
  const int bn   = blockIdx.x;
  const int wave = threadIdx.x >> 5;
  const int lane = threadIdx.x & 31;
  const bf16* ktb = kt + (size_t)bn * Dd * Ss;
  const bf16* vtb = vt + (size_t)bn * Dd * Ss;
  const int m0 = wave * 16;

  v8f acc[4] = {};
  stage_btile(vtb, Ss, lb[0]);
  for (int s0 = 0; s0 < Ss; s0 += 32) {
    const int cur = (s0 >> 5) & 1;
    wait_async();
    __syncthreads();
    if (s0 + 32 < Ss) stage_btile(vtb + s0 + 32, Ss, lb[cur ^ 1]);
    __builtin_prefetch(ktb + (size_t)m0 * Ss + s0 + 64, 0, 1);
    v16bf a = frag_a(ktb + (size_t)m0 * Ss + s0, Ss);   // rows = k-dim
    mma_group(a, lb[cur], acc);                          // cols = v-dim
    __syncthreads();
  }
  float* out = kv + (size_t)bn * Dd * Dd;
  const int rbase = m0 + (lane >> 4) * 8;
  const int cidx  = lane & 15;
#pragma unroll
  for (int nt = 0; nt < 4; ++nt)
#pragma unroll
    for (int i = 0; i < 8; ++i)
      out[(size_t)(rbase + i) * Dd + nt * 16 + cidx] = acc[nt][i];
}

// ----------------------- k_sum[b,n,d] = sum_s K^T --------------------------
__global__ void la_ksum(const bf16* __restrict__ kt, float* __restrict__ ksum) {
  const int t = blockIdx.x * blockDim.x + threadIdx.x;   // bn*64 + d
  const bf16* p = kt + (size_t)t * Ss;
  float s = 0.f;
#pragma unroll 8
  for (int i = 0; i < Ss; ++i) s += (float)p[i];
  ksum[t] = s;
}

// ------------- kv fp32 -> bf16, transposed (v-dim major) -------------------
__global__ void la_kv_cvt(const float* __restrict__ kv,
                          bf16* __restrict__ kvbT) {
  const int t  = blockIdx.x * blockDim.x + threadIdx.x;  // bn*4096 + r*64 + c
  const int bn = t >> 12;
  const int r  = (t >> 6) & 63;                          // k-dim
  const int c  = t & 63;                                 // v-dim
  kvbT[(size_t)bn * 4096 + (size_t)c * Dd + r] = (bf16)kv[t];
}

// ----------------------- den[b,s,n] = q . k_sum + eps ----------------------
__global__ void la_den(const bf16* __restrict__ qb,
                       const float* __restrict__ ksum,
                       float* __restrict__ den) {
  const int t   = blockIdx.x * blockDim.x + threadIdx.x; // tok*16 + n
  const int tok = t >> 4, n = t & 15;
  const int bn  = (tok >> 11) * Nh + n;
  const bf16*  q  = qb + (size_t)tok * Hh + n * Dd;
  const float* ks = ksum + (size_t)bn * Dd;
  float s = EPSc;
#pragma unroll 8
  for (int d = 0; d < Dd; ++d) s += (float)q[d] * ks[d];
  den[t] = s;
}

// ------- num = q @ kv (WMMA), fused epilogue -> combined (bf16) -----------
__global__ void la_attn(const bf16* __restrict__ qb,
                        const bf16* __restrict__ kvbT,
                        const float* __restrict__ den,
                        const float* __restrict__ gate,
                        const float* __restrict__ decay,
                        const float* __restrict__ gw,
                        bf16* __restrict__ comb) {
  __shared__ bf16 lkv[64 * 64];                  // 8 KB
  const int bn = blockIdx.y, b = bn >> 4, n = bn & 15;
  const int wave = threadIdx.x >> 5;
  const int lane = threadIdx.x & 31;
  const int m0   = blockIdx.x * 64 + wave * 16;  // s within batch

  const bf16* kvb = kvbT + (size_t)bn * 4096;
#pragma unroll
  for (int c = 0; c < 4; ++c) {
    const int ci = threadIdx.x + 128 * c;        // 0..511 chunks of 16B
    async_cp16(kvb + ci * 8, lkv + ci * 8);
  }

  // scalar memory-path coefficient (3 scales)
  const float e0 = __expf(gw[0]), e1 = __expf(gw[1]), e2 = __expf(gw[2]);
  const float cmem = (e0 * (1.f - sigmoidf_(decay[0])) +
                      e1 * (1.f - sigmoidf_(decay[1])) +
                      e2 * (1.f - sigmoidf_(decay[2]))) / (e0 + e1 + e2);

  const bf16* qbase = qb + (size_t)(b * Ss + m0) * Hh + n * Dd;
  wait_async();
  __syncthreads();

  v8f acc[4] = {};
#pragma unroll
  for (int k0 = 0; k0 < Dd; k0 += 32) {
    v16bf a = frag_a(qbase + k0, Hh);
    v16bf f[4];
#pragma unroll
    for (int nt = 0; nt < 4; ++nt)
      f[nt] = frag_b_lds(&lkv[nt * 16 * 64 + k0], 64);
#pragma unroll
    for (int nt = 0; nt < 4; ++nt) acc[nt] = wmma_bf16(a, f[nt], acc[nt]);
  }
  const int rbase = m0 + (lane >> 4) * 8;
  const int cidx  = lane & 15;
#pragma unroll
  for (int nt = 0; nt < 4; ++nt)
#pragma unroll
    for (int i = 0; i < 8; ++i) {
      const int tok = b * Ss + rbase + i;
      const float g = gate[tok];
      const float f = g / den[(size_t)tok * Nh + n] + (1.f - g) * cmem;
      comb[(size_t)tok * Hh + n * Dd + nt * 16 + cidx] = (bf16)(acc[nt][i] * f);
    }
}

// --------------------- out = combined @ Wo + bo (WMMA) ---------------------
__global__ void la_out_gemm(const bf16* __restrict__ comb,
                            const bf16* __restrict__ WoT,
                            const float* __restrict__ bo,
                            float* __restrict__ out) {
  __shared__ bf16 lb[2][64 * 32];                // 8 KB
  const int n0   = blockIdx.x * 64;
  const int wave = threadIdx.x >> 5;
  const int lane = threadIdx.x & 31;
  const int m0   = blockIdx.y * 64 + wave * 16;

  const bf16* wT   = WoT + (size_t)n0 * Hh;
  const bf16* arow = comb + (size_t)m0 * Hh;
  v8f acc[4] = {};

  stage_btile(wT, Hh, lb[0]);
  for (int k0 = 0; k0 < Hh; k0 += 32) {
    const int cur = (k0 >> 5) & 1;
    wait_async();
    __syncthreads();
    if (k0 + 32 < Hh) stage_btile(wT + k0 + 32, Hh, lb[cur ^ 1]);
    __builtin_prefetch(arow + k0 + 64, 0, 1);
    v16bf a = frag_a(arow + k0, Hh);
    mma_group(a, lb[cur], acc);
    __syncthreads();
  }
  const int rbase = m0 + (lane >> 4) * 8;
  const int cidx  = lane & 15;
#pragma unroll
  for (int nt = 0; nt < 4; ++nt) {
    const int col = n0 + nt * 16 + cidx;
#pragma unroll
    for (int i = 0; i < 8; ++i)
      out[(size_t)(rbase + i) * Hh + col] = acc[nt][i] + bo[col];
  }
}

// ---------------------------------------------------------------------------
extern "C" void kernel_launch(void* const* d_in, const int* in_sizes, int n_in,
                              void* d_out, int out_size, void* d_ws,
                              size_t ws_size, hipStream_t stream) {
  const float* x     = (const float*)d_in[0];
  const float* amask = (const float*)d_in[1];
  const float* Wq    = (const float*)d_in[2];
  const float* bq    = (const float*)d_in[3];
  const float* Wk    = (const float*)d_in[4];
  const float* bk    = (const float*)d_in[5];
  const float* Wv    = (const float*)d_in[6];
  const float* bv    = (const float*)d_in[7];
  const float* Wo    = (const float*)d_in[8];
  const float* bo    = (const float*)d_in[9];
  const float* Wg    = (const float*)d_in[10];
  const float* bg    = (const float*)d_in[11];
  const float* decay = (const float*)d_in[12];
  const float* gw    = (const float*)d_in[13];
  float* out = (float*)d_out;

  // -------- workspace carve (≈122 MB) --------
  char* p = (char*)d_ws;
  auto carve = [&](size_t bytes) { char* r = p; p += (bytes + 255) & ~size_t(255); return r; };
  bf16*  xb    = (bf16*)carve((size_t)TOK * Hh * 2);
  bf16*  Wcat  = (bf16*)carve((size_t)3 * HH2 * 2);    // [WqT|WkT|WvT]
  bf16*  WoT   = (bf16*)carve((size_t)HH2 * 2);
  bf16*  qb    = (bf16*)carve((size_t)TOK * Hh * 2);   // q (in-place RoPE'd)
  bf16*  kraw  = (bf16*)carve((size_t)TOK * Hh * 2);
  bf16*  vraw  = (bf16*)carve((size_t)TOK * Hh * 2);
  bf16*  kt    = (bf16*)carve((size_t)TOK * Hh * 2);   // [b,n][d][s]
  bf16*  vt    = (bf16*)carve((size_t)TOK * Hh * 2);   // [b,n][d][s]
  bf16*  comb  = (bf16*)carve((size_t)TOK * Hh * 2);
  float* kv    = (float*)carve((size_t)Bb * Nh * Dd * Dd * 4);
  bf16*  kvbT  = (bf16*)carve((size_t)Bb * Nh * Dd * Dd * 2);
  float* ksum  = (float*)carve((size_t)Bb * Nh * Dd * 4);
  float* den   = (float*)carve((size_t)TOK * Nh * 4);
  float* gate  = (float*)carve((size_t)TOK * 4);

  la_cvt_f32_bf16<<<(TOK * Hh) / 256, 256, 0, stream>>>(x, xb, TOK * Hh);
  la_cvt_transpose_w<<<HH2 / 256, 256, 0, stream>>>(Wq, Wcat);
  la_cvt_transpose_w<<<HH2 / 256, 256, 0, stream>>>(Wk, Wcat + (size_t)HH2);
  la_cvt_transpose_w<<<HH2 / 256, 256, 0, stream>>>(Wv, Wcat + (size_t)2 * HH2);
  la_cvt_transpose_w<<<HH2 / 256, 256, 0, stream>>>(Wo, WoT);

  la_qkv_proj<<<dim3(Hh / 64, TOK / 64), 128, 0, stream>>>(
      xb, Wcat, bq, bk, bv, qb, kraw, vraw);

  la_rope_mask<<<(TOK * Nh * 32) / 256, 256, 0, stream>>>(
      qb, kraw, vraw, kt, vt, amask);

  la_gate<<<TOK / 8, 256, 0, stream>>>(x, Wg, bg, gate);

  la_kv<<<Bb * Nh, 128, 0, stream>>>(kt, vt, kv);

  la_ksum<<<(Bb * Nh * Dd) / 256, 256, 0, stream>>>(kt, ksum);
  la_kv_cvt<<<(Bb * Nh * Dd * Dd) / 256, 256, 0, stream>>>(kv, kvbT);
  la_den<<<(TOK * Nh) / 256, 256, 0, stream>>>(qb, ksum, den);

  la_attn<<<dim3(Ss / 64, Bb * Nh), 128, 0, stream>>>(
      qb, kvbT, den, gate, decay, gw, comb);

  la_out_gemm<<<dim3(Hh / 64, TOK / 64), 128, 0, stream>>>(comb, WoT, bo, out);

  (void)in_sizes; (void)n_in; (void)out_size; (void)ws_size;
}
